// GNNWithAttention_18433999634685
// MI455X (gfx1250) — matrix operand order repaired and checked by reading, not compiled, not run on hardware
//
#include <hip/hip_runtime.h>
#include <hip/hip_bf16.h>
#include <math.h>

typedef __attribute__((ext_vector_type(16))) _Float16 v16h;
typedef __attribute__((ext_vector_type(8)))  _Float16 v8h;
typedef __attribute__((ext_vector_type(8)))  float    v8f;

#define NN     50000
#define EE     800000
#define EP     (EE + NN)     // edges + self loops
#define DM     128           // hidden*heads == input dim
#define HEADS  2
#define HID    64
#define GG     64
#define OUTC   32
#define EPSF   1e-5f
#define SLOPE  0.2f

#define MTILES   (NN / 16)   // 3125
#define LDS_STR  136         // padded LDS row stride in halves (4-bank rotate/row)

// ---------------------------------------------------------------------------
// f32 -> f16 conversion and weight transpose (for contiguous WMMA B loads)
// ---------------------------------------------------------------------------
__global__ void k_f32_to_f16(const float* __restrict__ in, _Float16* __restrict__ out, size_t n) {
    size_t i = (size_t)blockIdx.x * blockDim.x + threadIdx.x;
    if (i < n) out[i] = (_Float16)in[i];
}

// Wt[c][k] = W[k][c]  (W is [K=128][C=128] row-major)
__global__ void k_transpose_f16(const float* __restrict__ W, _Float16* __restrict__ Wt) {
    int i = blockIdx.x * blockDim.x + threadIdx.x;   // 128*128
    if (i < DM * DM) {
        int r = i >> 7, c = i & (DM - 1);
        Wt[c * DM + r] = (_Float16)W[r * DM + c];
    }
}

// ---------------------------------------------------------------------------
// WMMA GEMM: C[M x 128] = A[M x 128] (f16 row major) * W  (via Wt f16 [128 x 128])
// - Whole Bt is staged in LDS once per block (async-to-LDS when available),
//   padded rows for conflict-free column reads.
// - One wave32 computes a full 16x128 row block: 8 accumulators, 32 WMMA ops.
// ---------------------------------------------------------------------------
#if __has_builtin(__builtin_amdgcn_global_load_async_to_lds_b128) && \
    __has_builtin(__builtin_amdgcn_s_wait_asynccnt)
#define USE_ASYNC_LDS 1
typedef int v4i_n __attribute__((vector_size(16)));              // matches builtin pointee
typedef __attribute__((address_space(1))) v4i_n gas_v4i;         // prints as "__device__"
typedef __attribute__((address_space(3))) v4i_n las_v4i;         // prints as "__shared__"
#else
#define USE_ASYNC_LDS 0
#endif

__global__ __launch_bounds__(256)
void k_gemm_wmma(const _Float16* __restrict__ A, const _Float16* __restrict__ Bt,
                 float* __restrict__ C) {
    __shared__ _Float16 sB[DM * LDS_STR];

    int t = threadIdx.x;
    // ---- stage Bt (128x128 halves) into LDS, 8 halves (16B) per thread/iter ----
    #pragma unroll
    for (int i = 0; i < (DM * DM) / (256 * 8); ++i) {     // 8 iterations
        int c      = t + i * 256;                         // 16B chunk id
        int row    = c >> 4;
        int within = (c & 15) * 8;
#if USE_ASYNC_LDS
        gas_v4i* gp = (gas_v4i*)(unsigned long long)(Bt + (size_t)row * DM + within);
        las_v4i* lp = (las_v4i*)(unsigned)(unsigned long long)(&sB[row * LDS_STR + within]);
        __builtin_amdgcn_global_load_async_to_lds_b128(gp, lp, 0, 0);
#else
        *(v8h*)&sB[row * LDS_STR + within] = *(const v8h*)(Bt + (size_t)row * DM + within);
#endif
    }
#if USE_ASYNC_LDS
    __builtin_amdgcn_s_wait_asynccnt(0);
#endif
    __syncthreads();

    int lane = t & 31;
    int wave = blockIdx.x * (blockDim.x >> 5) + (t >> 5);
    if (wave < MTILES) {                       // wave-uniform: EXEC all-ones inside
        int hrow = lane & 15;                  // A row within tile / B col within tile
        int kh   = lane >> 4;                  // lane half -> K sub-range
        const _Float16* arow = A + (size_t)(wave * 16 + hrow) * DM;

        v8f acc[8];
        #pragma unroll
        for (int tn = 0; tn < 8; ++tn) acc[tn] = (v8f){};

        #pragma unroll
        for (int k0 = 0; k0 < DM; k0 += 32) {
            // A fragment (16x32 f16): elem e -> K=k0+16*kh+e
            v8h a0 = *(const v8h*)(arow + k0 + kh * 16);
            v8h a1 = *(const v8h*)(arow + k0 + kh * 16 + 8);
            v16h a;
            #pragma unroll
            for (int i = 0; i < 8; ++i) { a[i] = a0[i]; a[8 + i] = a1[i]; }
            // 8 N-tiles share this A fragment
            #pragma unroll
            for (int tn = 0; tn < 8; ++tn) {
                v16h b = *(const v16h*)&sB[(tn * 16 + hrow) * LDS_STR + k0 + kh * 16];
                acc[tn] = __builtin_amdgcn_wmma_f32_16x16x32_f16(false, a, false, b,
                                                                 (short)0, acc[tn],
                                                                 false, false);
            }
        }
        // C/D layout: VGPR r -> M = r (lanes 0-15) / r+8 (lanes 16-31), N = lane&15
        int cn0 = lane & 15;
        int cm0 = wave * 16 + ((lane < 16) ? 0 : 8);
        #pragma unroll
        for (int tn = 0; tn < 8; ++tn)
            #pragma unroll
            for (int r = 0; r < 8; ++r)
                C[(size_t)(cm0 + r) * DM + tn * 16 + cn0] = acc[tn][r];
    }
}

// ---------------------------------------------------------------------------
// Attention coefficients: a_src[n,h] = sum_c h[n,h,c]*att_src[h,c]  (flat c==t)
// ---------------------------------------------------------------------------
__global__ void k_attn_coef(const float* __restrict__ h, const float* __restrict__ asv,
                            const float* __restrict__ adv,
                            float* __restrict__ osrc, float* __restrict__ odst) {
    __shared__ float ss[DM], sd[DM];
    int n = blockIdx.x, t = threadIdx.x;
    float v = h[(size_t)n * DM + t];
    ss[t] = v * asv[t];
    sd[t] = v * adv[t];
    __syncthreads();
    for (int s = HID / 2; s > 0; s >>= 1) {
        if ((t & (HID - 1)) < s) { ss[t] += ss[t + s]; sd[t] += sd[t + s]; }
        __syncthreads();
    }
    if ((t & (HID - 1)) == 0) {
        int hd = t >> 6;
        osrc[n * HEADS + hd] = ss[t];
        odst[n * HEADS + hd] = sd[t];
    }
}

// ---------------------------------------------------------------------------
// fills
// ---------------------------------------------------------------------------
__global__ void k_fill_bias(float* __restrict__ agg, const float* __restrict__ b, size_t total) {
    size_t i = (size_t)blockIdx.x * blockDim.x + threadIdx.x;
    if (i < total) agg[i] = b[i & (DM - 1)];
}
__global__ void k_fill_attn_init(float* __restrict__ smax, float* __restrict__ ssum, int total) {
    int i = blockIdx.x * blockDim.x + threadIdx.x;
    if (i < total) { smax[i] = -INFINITY; ssum[i] = 0.f; }
}
__global__ void k_fill_zero(float* __restrict__ p, size_t n) {
    size_t i = (size_t)blockIdx.x * blockDim.x + threadIdx.x;
    if (i < n) p[i] = 0.f;
}

// ---------------------------------------------------------------------------
// edge passes
// ---------------------------------------------------------------------------
__device__ inline void atomicMaxF(float* addr, float v) {
    int* ai = (int*)addr;
    int old = __float_as_int(*addr);
    while (__int_as_float(old) < v) {
        int assumed = old;
        old = atomicCAS(ai, assumed, __float_as_int(v));
        if (old == assumed) break;
    }
}
__device__ inline float leakyf(float x) { return (x >= 0.f) ? x : SLOPE * x; }
__device__ inline float eluf(float x)   { return (x > 0.f) ? x : (expf(x) - 1.f); }

__global__ void k_edge_max(const int* __restrict__ ei, const float* __restrict__ asrc,
                           const float* __restrict__ adst, float* __restrict__ smax) {
    int i = blockIdx.x * blockDim.x + threadIdx.x;
    if (i >= EP * HEADS) return;
    int e = i >> 1, hd = i & 1;
    int s = (e < EE) ? ei[e] : (e - EE);
    int d = (e < EE) ? ei[EE + e] : (e - EE);
    float x = leakyf(asrc[s * HEADS + hd] + adst[d * HEADS + hd]);
    atomicMaxF(&smax[d * HEADS + hd], x);
}

__global__ void k_edge_expsum(const int* __restrict__ ei, const float* __restrict__ asrc,
                              const float* __restrict__ adst, const float* __restrict__ smax,
                              float* __restrict__ eebuf, float* __restrict__ ssum) {
    int i = blockIdx.x * blockDim.x + threadIdx.x;
    if (i >= EP * HEADS) return;
    int e = i >> 1, hd = i & 1;
    int s = (e < EE) ? ei[e] : (e - EE);
    int d = (e < EE) ? ei[EE + e] : (e - EE);
    float x = leakyf(asrc[s * HEADS + hd] + adst[d * HEADS + hd]);
    float v = expf(x - smax[d * HEADS + hd]);
    eebuf[(size_t)e * HEADS + hd] = v;
    atomicAdd(&ssum[d * HEADS + hd], v);
}

__global__ void k_edge_aggregate(const int* __restrict__ ei, const float* __restrict__ h,
                                 const float* __restrict__ eebuf, const float* __restrict__ ssum,
                                 float* __restrict__ agg) {
    long long i = (long long)blockIdx.x * blockDim.x + threadIdx.x;
    if (i >= (long long)EP * DM) return;
    long long e = i >> 7;
    int c  = (int)(i & (DM - 1));
    int hd = c >> 6;
    int s = (e < EE) ? ei[e] : (int)(e - EE);
    int d = (e < EE) ? ei[EE + e] : (int)(e - EE);
    float alpha = eebuf[e * HEADS + hd] / ssum[d * HEADS + hd];
    atomicAdd(&agg[(size_t)d * DM + c], h[(size_t)s * DM + c] * alpha);
}

// ---------------------------------------------------------------------------
// layer norms
// ---------------------------------------------------------------------------
__global__ void k_ln_elu_skip(const float* __restrict__ agg, const float* __restrict__ g,
                              const float* __restrict__ b, const float* __restrict__ skip,
                              const float* __restrict__ bskip, float* __restrict__ x1) {
    __shared__ float red[DM];
    int n = blockIdx.x, t = threadIdx.x;
    float v = agg[(size_t)n * DM + t];
    red[t] = v; __syncthreads();
    for (int s = DM / 2; s > 0; s >>= 1) { if (t < s) red[t] += red[t + s]; __syncthreads(); }
    float mean = red[0] * (1.f / DM);
    __syncthreads();
    float dv = v - mean;
    red[t] = dv * dv; __syncthreads();
    for (int s = DM / 2; s > 0; s >>= 1) { if (t < s) red[t] += red[t + s]; __syncthreads(); }
    float var = red[0] * (1.f / DM);
    float y = dv * rsqrtf(var + EPSF) * g[t] + b[t];
    x1[(size_t)n * DM + t] = eluf(y) + skip[(size_t)n * DM + t] + bskip[t];
}

__global__ void k_add_ln_elu(const float* __restrict__ agg, const float* __restrict__ x1,
                             const float* __restrict__ g, const float* __restrict__ b,
                             float* __restrict__ emb) {
    __shared__ float red[DM];
    int n = blockIdx.x, t = threadIdx.x;
    float v = agg[(size_t)n * DM + t] + x1[(size_t)n * DM + t];
    red[t] = v; __syncthreads();
    for (int s = DM / 2; s > 0; s >>= 1) { if (t < s) red[t] += red[t + s]; __syncthreads(); }
    float mean = red[0] * (1.f / DM);
    __syncthreads();
    float dv = v - mean;
    red[t] = dv * dv; __syncthreads();
    for (int s = DM / 2; s > 0; s >>= 1) { if (t < s) red[t] += red[t + s]; __syncthreads(); }
    float var = red[0] * (1.f / DM);
    float y = dv * rsqrtf(var + EPSF) * g[t] + b[t];
    emb[(size_t)n * DM + t] = eluf(y);
}

// ---------------------------------------------------------------------------
// pooling + head
// ---------------------------------------------------------------------------
__global__ void k_pool_count(const int* __restrict__ batch, float* __restrict__ cnt) {
    int n = blockIdx.x * blockDim.x + threadIdx.x;
    if (n < NN) atomicAdd(&cnt[batch[n]], 1.f);
}
__global__ void k_pool_sum(const float* __restrict__ emb, const int* __restrict__ batch,
                           float* __restrict__ pool) {
    size_t i = (size_t)blockIdx.x * blockDim.x + threadIdx.x;
    if (i < (size_t)NN * DM) {
        int n = (int)(i >> 7), c = (int)(i & (DM - 1));
        atomicAdd(&pool[batch[n] * DM + c], emb[i]);
    }
}
__global__ void k_fc(const float* __restrict__ pool, const float* __restrict__ cnt,
                     const float* __restrict__ Wfc, const float* __restrict__ bfc,
                     float* __restrict__ logits) {
    int i = blockIdx.x * blockDim.x + threadIdx.x;
    if (i < GG * OUTC) {
        int gidx = i / OUTC, o = i % OUTC;
        float inv = 1.f / fmaxf(cnt[gidx], 1.f);
        float acc = bfc[o];
        for (int c = 0; c < DM; ++c)
            acc += pool[gidx * DM + c] * inv * Wfc[c * OUTC + o];
        logits[i] = acc;
    }
}
__global__ void k_bn(const float* __restrict__ logits, const float* __restrict__ g,
                     const float* __restrict__ b, float* __restrict__ out) {
    int o = threadIdx.x;   // 32 threads (one wave)
    if (o >= OUTC) return;
    float m = 0.f;
    for (int r = 0; r < GG; ++r) m += logits[r * OUTC + o];
    m *= (1.f / GG);
    float v = 0.f;
    for (int r = 0; r < GG; ++r) { float d = logits[r * OUTC + o] - m; v += d * d; }
    v *= (1.f / GG);
    float s = rsqrtf(v + EPSF) * g[o];
    float bb = b[o];
    for (int r = 0; r < GG; ++r)
        out[r * OUTC + o] = (logits[r * OUTC + o] - m) * s + bb;
}

// ---------------------------------------------------------------------------
extern "C" void kernel_launch(void* const* d_in, const int* in_sizes, int n_in,
                              void* d_out, int out_size, void* d_ws, size_t ws_size,
                              hipStream_t stream) {
    const float* x    = (const float*)d_in[0];
    const int*   ei   = (const int*)d_in[1];
    const int*   batch= (const int*)d_in[2];
    const float* W1   = (const float*)d_in[3];
    const float* as1  = (const float*)d_in[4];
    const float* ad1  = (const float*)d_in[5];
    const float* b1   = (const float*)d_in[6];
    const float* ln1g = (const float*)d_in[7];
    const float* ln1b = (const float*)d_in[8];
    const float* Wsk  = (const float*)d_in[9];
    const float* bsk  = (const float*)d_in[10];
    const float* W2   = (const float*)d_in[11];
    const float* as2  = (const float*)d_in[12];
    const float* ad2  = (const float*)d_in[13];
    const float* b2   = (const float*)d_in[14];
    const float* ln2g = (const float*)d_in[15];
    const float* ln2b = (const float*)d_in[16];
    const float* Wfc  = (const float*)d_in[17];
    const float* bfc  = (const float*)d_in[18];
    const float* bng  = (const float*)d_in[19];
    const float* bnb  = (const float*)d_in[20];
    float* out = (float*)d_out;

    char* wsb = (char*)d_ws;
    size_t off = 0;
    auto take = [&](size_t bytes) -> char* {
        char* p = wsb + off;
        off += (bytes + 255) & ~(size_t)255;
        return p;
    };
    float* h      = (float*)take((size_t)NN * DM * 4);
    float* agg    = (float*)take((size_t)NN * DM * 4);
    float* x1     = (float*)take((size_t)NN * DM * 4);
    float* emb    = (float*)take((size_t)NN * DM * 4);   // skip, then emb
    float* asrc   = (float*)take((size_t)NN * HEADS * 4);
    float* adst   = (float*)take((size_t)NN * HEADS * 4);
    float* smax   = (float*)take((size_t)NN * HEADS * 4);
    float* ssum   = (float*)take((size_t)NN * HEADS * 4);
    float* eebuf  = (float*)take((size_t)EP * HEADS * 4);
    float* pool   = (float*)take((size_t)GG * DM * 4);
    float* cnt    = (float*)take((size_t)GG * 4);
    float* logits = (float*)take((size_t)GG * OUTC * 4);
    _Float16* xh  = (_Float16*)take((size_t)NN * DM * 2);
    _Float16* wt1 = (_Float16*)take((size_t)DM * DM * 2);
    _Float16* wts = (_Float16*)take((size_t)DM * DM * 2);
    _Float16* wt2 = (_Float16*)take((size_t)DM * DM * 2);

    const int T = 256;
    size_t nd      = (size_t)NN * DM;
    int    bND     = (int)((nd + T - 1) / T);
    int    bWT     = (DM * DM + T - 1) / T;
    int    bEdge   = (EP * HEADS + T - 1) / T;
    int    bEAgg   = (int)(((long long)EP * DM + T - 1) / T);
    int    bGemm   = (MTILES + 7) / 8;            // 8 waves/block, 1 row-block/wave

    // --- weight prep ---
    k_transpose_f16<<<bWT, T, 0, stream>>>(W1,  wt1);
    k_transpose_f16<<<bWT, T, 0, stream>>>(Wsk, wts);
    k_transpose_f16<<<bWT, T, 0, stream>>>(W2,  wt2);

    // ===================== GAT layer 1 =====================
    k_f32_to_f16<<<bND, T, 0, stream>>>(x, xh, nd);
    k_gemm_wmma<<<bGemm, T, 0, stream>>>(xh, wt1, h);     // h1 = x @ W1
    k_gemm_wmma<<<bGemm, T, 0, stream>>>(xh, wts, emb);   // skip = x @ Wskip
    k_attn_coef<<<NN, DM, 0, stream>>>(h, as1, ad1, asrc, adst);
    k_fill_bias<<<bND, T, 0, stream>>>(agg, b1, nd);
    k_fill_attn_init<<<(NN * HEADS + T - 1) / T, T, 0, stream>>>(smax, ssum, NN * HEADS);
    k_edge_max<<<bEdge, T, 0, stream>>>(ei, asrc, adst, smax);
    k_edge_expsum<<<bEdge, T, 0, stream>>>(ei, asrc, adst, smax, eebuf, ssum);
    k_edge_aggregate<<<bEAgg, T, 0, stream>>>(ei, h, eebuf, ssum, agg);
    k_ln_elu_skip<<<NN, DM, 0, stream>>>(agg, ln1g, ln1b, emb, bsk, x1);

    // ===================== GAT layer 2 =====================
    k_f32_to_f16<<<bND, T, 0, stream>>>(x1, xh, nd);
    k_gemm_wmma<<<bGemm, T, 0, stream>>>(xh, wt2, h);     // h2 = x1 @ W2
    k_attn_coef<<<NN, DM, 0, stream>>>(h, as2, ad2, asrc, adst);
    k_fill_bias<<<bND, T, 0, stream>>>(agg, b2, nd);
    k_fill_attn_init<<<(NN * HEADS + T - 1) / T, T, 0, stream>>>(smax, ssum, NN * HEADS);
    k_edge_max<<<bEdge, T, 0, stream>>>(ei, asrc, adst, smax);
    k_edge_expsum<<<bEdge, T, 0, stream>>>(ei, asrc, adst, smax, eebuf, ssum);
    k_edge_aggregate<<<bEAgg, T, 0, stream>>>(ei, h, eebuf, ssum, agg);
    k_add_ln_elu<<<NN, DM, 0, stream>>>(agg, x1, ln2g, ln2b, emb);

    // ===================== pool + head =====================
    k_fill_zero<<<(GG * DM + T - 1) / T, T, 0, stream>>>(pool, (size_t)GG * DM);
    k_fill_zero<<<1, GG, 0, stream>>>(cnt, (size_t)GG);
    k_pool_count<<<(NN + T - 1) / T, T, 0, stream>>>(batch, cnt);
    k_pool_sum<<<bND, T, 0, stream>>>(emb, batch, pool);
    k_fc<<<(GG * OUTC + T - 1) / T, T, 0, stream>>>(pool, cnt, Wfc, bfc, logits);
    k_bn<<<1, 32, 0, stream>>>(logits, bng, bnb, out);
}
